// Qwen3VLMoeTextDecoderLayer_74457553043879
// MI455X (gfx1250) — matrix operand order repaired and checked by reading, not compiled, not run on hardware
//
#include <hip/hip_runtime.h>
#include <hip/hip_bf16.h>
#include <cstdint>

// ---------------- problem dims (Qwen3-VL MoE decoder layer) ----------------
#define BB   2
#define SS   1024
#define HH   2048
#define NHq  16
#define NKVh 4
#define HDd  128
#define EE   32
#define TOPK 8
#define FF   768
#define TT   (BB * SS)      // 2048 tokens
#define TKR  (TT * TOPK)    // 16384 routed rows

// ---------------- WMMA types / helpers ----------------
typedef __attribute__((ext_vector_type(16))) __bf16   v16bf;
typedef __attribute__((ext_vector_type(8)))  float    v8f;
typedef __attribute__((ext_vector_type(8)))  unsigned v8u;

static __device__ __forceinline__ unsigned short f2bf(float f) {
  unsigned u = __builtin_bit_cast(unsigned, f);
  u += 0x7fffu + ((u >> 16) & 1u);        // round-to-nearest-even
  return (unsigned short)(u >> 16);
}
static __device__ __forceinline__ v8f vzero8() {
  v8f z = {0.f, 0.f, 0.f, 0.f, 0.f, 0.f, 0.f, 0.f};
  return z;
}
static __device__ __forceinline__ v8f wmma_bf16(v16bf a, v16bf b, v8f c) {
  return __builtin_amdgcn_wmma_f32_16x16x32_bf16(false, a, false, b, (short)0, c, false, false);
}
// A fragment (16x32 bf16, row-major source): lane L -> row=L%16,
// VGPR v -> K = (L/16)*16 + 2v (+0/1 packed). One 32B run.
static __device__ __forceinline__ v16bf load_a_bf16(const unsigned short* A, int lda,
                                                    int row, int k0, int half) {
  const unsigned short* base = A + (size_t)row * lda + k0;
  v8u w;
#pragma unroll
  for (int v = 0; v < 8; ++v) {
    int kk = half * 16 + (v << 1);
    w[v] = *(const unsigned*)(base + kk);
  }
  return __builtin_bit_cast(v16bf, w);
}
// B fragment (32x16 bf16) with K contiguous per lane (N-major / transposed storage):
// lane L -> col=L%16, VGPR v -> K = (L/16)*16 + 2v (+0/1). One 32B run.
static __device__ __forceinline__ v16bf load_b_k16(const unsigned short* base, int half) {
  v8u w;
#pragma unroll
  for (int v = 0; v < 8; ++v)
    w[v] = *(const unsigned*)(base + half * 16 + (v << 1));
  return __builtin_bit_cast(v16bf, w);
}

// ---------------- batched transpose + fp32->bf16 convert ----------------
// in[z][r][c] (fp32, row-major) -> out[z][c][r] (bf16). grid=(C/32, R/32, Z), block=256
__global__ void __launch_bounds__(256)
transpose_cvt_bf16(const float* __restrict__ in, unsigned short* __restrict__ outp,
                   int R, int C) {
  __shared__ float tile[32][33];
  int z = blockIdx.z;
  const float* inz = in + (size_t)z * R * C;
  unsigned short* outz = outp + (size_t)z * R * C;
  int c0 = blockIdx.x * 32, r0 = blockIdx.y * 32;
  int tx = threadIdx.x & 31, ty = threadIdx.x >> 5;
#pragma unroll
  for (int k = 0; k < 32; k += 8) {
    int r = r0 + ty + k, c = c0 + tx;
    tile[ty + k][tx] = (r < R && c < C) ? inz[(size_t)r * C + c] : 0.f;
  }
  __syncthreads();
#pragma unroll
  for (int k = 0; k < 32; k += 8) {
    int c = c0 + ty + k, r = r0 + tx;
    if (c < C && r < R) outz[(size_t)c * R + r] = f2bf(tile[tx][ty + k]);
  }
}

// V: [B,S,NKV,HD] fp32 -> vt[B,NKV,HD,S] bf16.  grid=(S/32, HD/32, B*NKV)
__global__ void __launch_bounds__(256)
v_transpose(const float* __restrict__ vf, unsigned short* __restrict__ vt) {
  __shared__ float tile[32][33];
  int zb = blockIdx.z;
  int b = zb / NKVh, kvh = zb % NKVh;
  int s0 = blockIdx.x * 32, d0 = blockIdx.y * 32;
  int tx = threadIdx.x & 31, ty = threadIdx.x >> 5;
#pragma unroll
  for (int k = 0; k < 32; k += 8) {
    int s = s0 + ty + k, d = d0 + tx;
    tile[ty + k][tx] = vf[((size_t)(b * SS + s) * NKVh + kvh) * HDd + d];
  }
  __syncthreads();
  unsigned short* outz = vt + (size_t)zb * HDd * SS;
#pragma unroll
  for (int k = 0; k < 32; k += 8) {
    int d = d0 + ty + k, s = s0 + tx;
    outz[(size_t)d * SS + s] = f2bf(tile[tx][ty + k]);
  }
}

// ---------------- WMMA GEMM: C[M,N] = A(bf16,[M,K]) * B(bf16, N-major [N,K]) ----------------
// one wave computes a 64x64 tile; K multiple of 32.
__global__ void __launch_bounds__(32)
gemm_bf16(const unsigned short* __restrict__ A, const unsigned short* __restrict__ Bt,
          float* __restrict__ C, int M, int N, int Kd) {
  int lane = threadIdx.x, half = lane >> 4, l16 = lane & 15;
  int n0 = blockIdx.x * 64, m0 = blockIdx.y * 64;
  v8f acc[4][4];
#pragma unroll
  for (int i = 0; i < 4; ++i)
#pragma unroll
    for (int j = 0; j < 4; ++j) acc[i][j] = vzero8();
  int arow[4];
  const unsigned short* bbase[4];
#pragma unroll
  for (int i = 0; i < 4; ++i) {
    int r = m0 + i * 16 + l16; arow[i] = r < M ? r : M - 1;
    int c = n0 + i * 16 + l16; bbase[i] = Bt + (size_t)(c < N ? c : N - 1) * Kd;
  }
  for (int k0 = 0; k0 < Kd; k0 += 32) {
    v16bf af[4], bfv[4];
#pragma unroll
    for (int i = 0; i < 4; ++i) af[i] = load_a_bf16(A, Kd, arow[i], k0, half);
#pragma unroll
    for (int j = 0; j < 4; ++j) bfv[j] = load_b_k16(bbase[j] + k0, half);
#pragma unroll
    for (int i = 0; i < 4; ++i)
#pragma unroll
      for (int j = 0; j < 4; ++j)
        acc[i][j] = wmma_bf16(af[i], bfv[j], acc[i][j]);
  }
#pragma unroll
  for (int i = 0; i < 4; ++i)
#pragma unroll
    for (int j = 0; j < 4; ++j)
#pragma unroll
      for (int r = 0; r < 8; ++r) {
        int row = m0 + i * 16 + half * 8 + r;
        int col = n0 + j * 16 + l16;
        if (row < M && col < N) C[(size_t)row * N + col] = acc[i][j][r];
      }
}

// ---------------- fused MoE up-projection: act = silu(A*WiT) * (A*WuT), bf16 out ----------
// A rows gathered through rowtok (slot -> token) from h2b. 64x32 dual-output tile per wave.
__global__ void __launch_bounds__(32)
moe_up_swiglu(const unsigned short* __restrict__ h2b, const unsigned short* __restrict__ WiT,
              const unsigned short* __restrict__ WuT, unsigned short* __restrict__ act,
              const int* __restrict__ offs, const int* __restrict__ rowtok) {
  int e = blockIdx.z;
  int g0 = offs[e], size = offs[e + 1] - g0;
  int m0 = blockIdx.y * 64;
  if (m0 >= size) return;
  int lane = threadIdx.x, half = lane >> 4, l16 = lane & 15;
  int n0 = blockIdx.x * 32;
  const unsigned short* Bi = WiT + (size_t)e * HH * FF;
  const unsigned short* Bu = WuT + (size_t)e * HH * FF;
  v8f ag[4][2], au[4][2];
#pragma unroll
  for (int i = 0; i < 4; ++i)
#pragma unroll
    for (int j = 0; j < 2; ++j) { ag[i][j] = vzero8(); au[i][j] = vzero8(); }
  int arow[4];
#pragma unroll
  for (int i = 0; i < 4; ++i) {
    int r = m0 + i * 16 + l16;
    arow[i] = rowtok[g0 + (r < size ? r : size - 1)];
  }
  const unsigned short* bib[2];
  const unsigned short* bub[2];
#pragma unroll
  for (int j = 0; j < 2; ++j) {
    int c = n0 + j * 16 + l16;
    bib[j] = Bi + (size_t)c * HH;
    bub[j] = Bu + (size_t)c * HH;
  }
  for (int k0 = 0; k0 < HH; k0 += 32) {
    v16bf af[4], bi[2], bu[2];
#pragma unroll
    for (int i = 0; i < 4; ++i) af[i] = load_a_bf16(h2b, HH, arow[i], k0, half);
#pragma unroll
    for (int j = 0; j < 2; ++j) { bi[j] = load_b_k16(bib[j] + k0, half); bu[j] = load_b_k16(bub[j] + k0, half); }
#pragma unroll
    for (int i = 0; i < 4; ++i)
#pragma unroll
      for (int j = 0; j < 2; ++j) {
        ag[i][j] = wmma_bf16(af[i], bi[j], ag[i][j]);
        au[i][j] = wmma_bf16(af[i], bu[j], au[i][j]);
      }
  }
#pragma unroll
  for (int i = 0; i < 4; ++i)
#pragma unroll
    for (int j = 0; j < 2; ++j)
#pragma unroll
      for (int r = 0; r < 8; ++r) {
        int row = m0 + i * 16 + half * 8 + r;
        int col = n0 + j * 16 + l16;
        if (row < size) {
          float g = ag[i][j][r], u = au[i][j][r];
          float sv = g / (1.f + __expf(-g));  // silu
          act[(size_t)(g0 + row) * FF + col] = f2bf(sv * u);
        }
      }
}

// ---------------- grouped down-projection: y = act * WdT (per expert) ----------------
__global__ void __launch_bounds__(32)
gemm_moe_down(const unsigned short* __restrict__ A, const unsigned short* __restrict__ Ballt,
              float* __restrict__ C, const int* __restrict__ offs, int Kd, int N) {
  int e = blockIdx.z;
  int g0 = offs[e], size = offs[e + 1] - g0;
  int m0 = blockIdx.y * 64;
  if (m0 >= size) return;
  const unsigned short* Bt = Ballt + (size_t)e * Kd * N;
  int lane = threadIdx.x, half = lane >> 4, l16 = lane & 15;
  int n0 = blockIdx.x * 64;
  v8f acc[4][4];
#pragma unroll
  for (int i = 0; i < 4; ++i)
#pragma unroll
    for (int j = 0; j < 4; ++j) acc[i][j] = vzero8();
  int arow[4];
  const unsigned short* bbase[4];
#pragma unroll
  for (int i = 0; i < 4; ++i) {
    int r = m0 + i * 16 + l16;
    arow[i] = g0 + (r < size ? r : size - 1);
    int c = n0 + i * 16 + l16;
    bbase[i] = Bt + (size_t)(c < N ? c : N - 1) * Kd;
  }
  for (int k0 = 0; k0 < Kd; k0 += 32) {
    v16bf af[4], bfv[4];
#pragma unroll
    for (int i = 0; i < 4; ++i) af[i] = load_a_bf16(A, Kd, arow[i], k0, half);
#pragma unroll
    for (int j = 0; j < 4; ++j) bfv[j] = load_b_k16(bbase[j] + k0, half);
#pragma unroll
    for (int i = 0; i < 4; ++i)
#pragma unroll
      for (int j = 0; j < 4; ++j)
        acc[i][j] = wmma_bf16(af[i], bfv[j], acc[i][j]);
  }
#pragma unroll
  for (int i = 0; i < 4; ++i)
#pragma unroll
    for (int j = 0; j < 4; ++j)
#pragma unroll
      for (int r = 0; r < 8; ++r) {
        int row = m0 + i * 16 + half * 8 + r;
        int col = n0 + j * 16 + l16;
        if (row < size && col < N) C[(size_t)(g0 + row) * N + col] = acc[i][j][r];
      }
}

// ---------------- flash attention (WMMA), 16 queries per wave ----------------
__global__ void __launch_bounds__(32)
attn_fwd(const unsigned short* __restrict__ qb, const unsigned short* __restrict__ kb,
         const unsigned short* __restrict__ vt, unsigned short* __restrict__ ob) {
  __shared__ __align__(16) unsigned short pmat[16 * 32];
  const float scale = 0.088388347648318447f;  // 1/sqrt(128)
  int lane = threadIdx.x, half = lane >> 4, l16 = lane & 15;
  int qbase = blockIdx.x * 16;
  int h = blockIdx.y;
  int b = blockIdx.z;
  int kvh = h / (NHq / NKVh);
  const unsigned short* Qh = qb + ((size_t)b * SS * NHq + h) * HDd;
  const unsigned short* Kh = kb + ((size_t)b * SS * NKVh + kvh) * HDd;
  const unsigned short* Vt = vt + (size_t)(b * NKVh + kvh) * HDd * SS;

  v16bf qa[4];
#pragma unroll
  for (int c = 0; c < 4; ++c)
    qa[c] = load_a_bf16(Qh, NHq * HDd, qbase + l16, c * 32, half);

  v8f O[8];
#pragma unroll
  for (int o = 0; o < 8; ++o) O[o] = vzero8();
  float mrow[8], lrow[8];
#pragma unroll
  for (int r = 0; r < 8; ++r) { mrow[r] = -1e30f; lrow[r] = 0.f; }

  for (int kbase = 0; kbase <= qbase + 15; kbase += 32) {
    v8f s0 = vzero8(), s1 = vzero8();
    const unsigned short* kc0 = Kh + (size_t)(kbase + l16) * (NKVh * HDd);
    const unsigned short* kc1 = Kh + (size_t)(kbase + 16 + l16) * (NKVh * HDd);
#pragma unroll
    for (int c = 0; c < 4; ++c) {
      v16bf b0 = load_b_k16(kc0 + c * 32, half);
      v16bf b1 = load_b_k16(kc1 + c * 32, half);
      s0 = wmma_bf16(qa[c], b0, s0);
      s1 = wmma_bf16(qa[c], b1, s1);
    }
#pragma unroll
    for (int r = 0; r < 8; ++r) {
      int qrow = qbase + half * 8 + r;
      float v0 = (kbase + l16      <= qrow) ? s0[r] * scale : -1e30f;
      float v1 = (kbase + 16 + l16 <= qrow) ? s1[r] * scale : -1e30f;
      float rm = fmaxf(v0, v1);
#pragma unroll
      for (int m = 8; m >= 1; m >>= 1) rm = fmaxf(rm, __shfl_xor(rm, m, 32));
      float mnew = fmaxf(mrow[r], rm);
      float alpha = __expf(mrow[r] - mnew);
      mrow[r] = mnew;
      float p0 = __expf(v0 - mnew);
      float p1 = __expf(v1 - mnew);
      float rs = p0 + p1;
#pragma unroll
      for (int m = 8; m >= 1; m >>= 1) rs += __shfl_xor(rs, m, 32);
      lrow[r] = lrow[r] * alpha + rs;
      pmat[(half * 8 + r) * 32 + l16] = f2bf(p0);
      pmat[(half * 8 + r) * 32 + 16 + l16] = f2bf(p1);
#pragma unroll
      for (int o = 0; o < 8; ++o) O[o][r] *= alpha;
    }
    __syncthreads();
    v16bf pa = load_a_bf16(pmat, 32, l16, 0, half);
#pragma unroll
    for (int o = 0; o < 8; ++o) {
      v16bf vf = load_b_k16(Vt + (size_t)(o * 16 + l16) * SS + kbase, half);
      O[o] = wmma_bf16(pa, vf, O[o]);
    }
    __syncthreads();
  }
#pragma unroll
  for (int o = 0; o < 8; ++o)
#pragma unroll
    for (int r = 0; r < 8; ++r) {
      int qrow = qbase + half * 8 + r;
      ob[((size_t)(b * SS + qrow)) * (NHq * HDd) + h * HDd + o * 16 + l16] =
          f2bf(O[o][r] / lrow[r]);
    }
}

// ---------------- elementwise / norm / router kernels ----------------
__global__ void __launch_bounds__(256)
rmsnorm_bf16(const float* __restrict__ x, const float* __restrict__ w,
             unsigned short* __restrict__ y, int Hdim) {
  __shared__ float red[8];
  int row = blockIdx.x, tid = threadIdx.x;
  const float* xr = x + (size_t)row * Hdim;
  float ss = 0.f;
  for (int i = tid; i < Hdim; i += 256) { float v = xr[i]; ss += v * v; }
#pragma unroll
  for (int m = 16; m >= 1; m >>= 1) ss += __shfl_xor(ss, m, 32);
  if ((tid & 31) == 0) red[tid >> 5] = ss;
  __syncthreads();
  float tot = 0.f;
#pragma unroll
  for (int i = 0; i < 8; ++i) tot += red[i];
  float rs = rsqrtf(tot / (float)Hdim + 1e-6f);
  unsigned short* yr = y + (size_t)row * Hdim;
  for (int i = tid; i < Hdim; i += 256) yr[i] = f2bf(xr[i] * rs * w[i]);
}

// per-head QK RMSNorm + RoPE -> bf16. grid=(T, NH+NKV), block=128
__global__ void __launch_bounds__(128)
qk_rope(const float* __restrict__ qf, const float* __restrict__ kf,
        const float* __restrict__ qw, const float* __restrict__ kw,
        unsigned short* __restrict__ qb, unsigned short* __restrict__ kb) {
  __shared__ float sh[HDd];
  __shared__ float red[4];
  int t = blockIdx.x, head = blockIdx.y, d = threadIdx.x;
  int s = t % SS;
  const float* src; const float* w; unsigned short* dst;
  if (head < NHq) {
    src = qf + ((size_t)t * NHq + head) * HDd; w = qw;
    dst = qb + ((size_t)t * NHq + head) * HDd;
  } else {
    int kh = head - NHq;
    src = kf + ((size_t)t * NKVh + kh) * HDd; w = kw;
    dst = kb + ((size_t)t * NKVh + kh) * HDd;
  }
  float x = src[d];
  float ss = x * x;
#pragma unroll
  for (int m = 16; m >= 1; m >>= 1) ss += __shfl_xor(ss, m, 32);
  if ((d & 31) == 0) red[d >> 5] = ss;
  __syncthreads();
  float tot = red[0] + red[1] + red[2] + red[3];
  float xn = x * rsqrtf(tot / (float)HDd + 1e-6f) * w[d];
  sh[d] = xn;
  __syncthreads();
  int i = d & 63;
  float inv = powf(1.0e6f, -((float)i) / 64.0f);
  float ang = (float)s * inv;
  float c = cosf(ang), sn = sinf(ang);
  float other = (d < 64) ? -sh[d + 64] : sh[d - 64];
  dst[d] = f2bf(xn * c + other * sn);
}

__global__ void __launch_bounds__(256)
addvec(const float* __restrict__ a, const float* __restrict__ b,
       float* __restrict__ c, int n) {
  int i = blockIdx.x * 256 + threadIdx.x;
  if (i < n) c[i] = a[i] + b[i];
}

__global__ void __launch_bounds__(64)
zero_i32(int* __restrict__ p, int n) {
  int i = threadIdx.x;
  if (i < n) p[i] = 0;
}

// one wave per token: softmax over 32 experts + top-8 (ties -> lowest index)
__global__ void __launch_bounds__(32)
router_topk(const float* __restrict__ logits, float* __restrict__ topw,
            int* __restrict__ topi, int* __restrict__ counts) {
  int t = blockIdx.x, e = threadIdx.x;
  float lg = logits[(size_t)t * EE + e];
  float mx = lg;
#pragma unroll
  for (int m = 16; m >= 1; m >>= 1) mx = fmaxf(mx, __shfl_xor(mx, m, 32));
  float p = __expf(lg - mx);
  float sm = p;
#pragma unroll
  for (int m = 16; m >= 1; m >>= 1) sm += __shfl_xor(sm, m, 32);
  float prob = p / sm;
  float pv = prob;
  float wsel[TOPK]; int isel[TOPK];
#pragma unroll
  for (int j = 0; j < TOPK; ++j) {
    float v = pv; int idx = e;
#pragma unroll
    for (int m = 16; m >= 1; m >>= 1) {
      float ov = __shfl_xor(v, m, 32);
      int oi = __shfl_xor(idx, m, 32);
      if (ov > v || (ov == v && oi < idx)) { v = ov; idx = oi; }
    }
    wsel[j] = v; isel[j] = idx;
    if (e == idx) pv = -1.f;
  }
  float wsum = 0.f;
#pragma unroll
  for (int j = 0; j < TOPK; ++j) wsum += wsel[j];
  if (e < TOPK) {
    topw[(size_t)t * TOPK + e] = wsel[e] / wsum;
    topi[(size_t)t * TOPK + e] = isel[e];
    atomicAdd(&counts[isel[e]], 1);
  }
}

__global__ void scan_offsets(const int* __restrict__ counts, int* __restrict__ offs,
                             int* __restrict__ cursor) {
  if (threadIdx.x == 0) {
    int acc = 0;
    for (int e = 0; e < EE; ++e) { offs[e] = acc; cursor[e] = acc; acc += counts[e]; }
    offs[EE] = acc;
  }
}

__global__ void __launch_bounds__(256)
assign_slots(const int* __restrict__ topi, int* __restrict__ cursor,
             int* __restrict__ slot_of, int* __restrict__ rowtok) {
  int idx = blockIdx.x * 256 + threadIdx.x;
  if (idx >= TKR) return;
  int e = topi[idx];
  int slot = atomicAdd(&cursor[e], 1);
  slot_of[idx] = slot;
  rowtok[slot] = idx >> 3;  // token id
}

// deterministic combine: fixed k order per token
__global__ void __launch_bounds__(256)
combine_moe(const float* __restrict__ x1, const float* __restrict__ y,
            const float* __restrict__ topw, const int* __restrict__ slot_of,
            float* __restrict__ out) {
  int t = blockIdx.x, tid = threadIdx.x;
  int slots[TOPK]; float wts[TOPK];
#pragma unroll
  for (int j = 0; j < TOPK; ++j) {
    slots[j] = slot_of[(size_t)t * TOPK + j];
    wts[j]   = topw[(size_t)t * TOPK + j];
  }
  for (int i = tid; i < HH; i += 256) {
    float s = x1[(size_t)t * HH + i];
#pragma unroll
    for (int j = 0; j < TOPK; ++j) s += wts[j] * y[(size_t)slots[j] * HH + i];
    out[(size_t)t * HH + i] = s;
  }
}

// ---------------- host-side orchestration ----------------
extern "C" void kernel_launch(void* const* d_in, const int* in_sizes, int n_in,
                              void* d_out, int out_size, void* d_ws, size_t ws_size,
                              hipStream_t stream) {
  (void)in_sizes; (void)n_in; (void)out_size; (void)ws_size;
  const float* x   = (const float*)d_in[0];
  const float* wq  = (const float*)d_in[1];
  const float* wk  = (const float*)d_in[2];
  const float* wv  = (const float*)d_in[3];
  const float* wo  = (const float*)d_in[4];
  const float* qnw = (const float*)d_in[5];
  const float* knw = (const float*)d_in[6];
  const float* ln1 = (const float*)d_in[7];
  const float* ln2 = (const float*)d_in[8];
  const float* gw  = (const float*)d_in[9];
  const float* wi  = (const float*)d_in[10];
  const float* wu  = (const float*)d_in[11];
  const float* wd  = (const float*)d_in[12];
  float* out = (float*)d_out;

  char* p = (char*)d_ws;
  auto alloc = [&](size_t bytes) -> char* {
    char* r = p;
    p += (bytes + 255) & ~(size_t)255;
    return r;
  };
  // activations
  unsigned short* h1b   = (unsigned short*)alloc((size_t)TT * HH * 2);
  float*          qf    = (float*)alloc((size_t)TT * NHq * HDd * 4);
  float*          kf    = (float*)alloc((size_t)TT * NKVh * HDd * 4);
  float*          vf    = (float*)alloc((size_t)TT * NKVh * HDd * 4);
  unsigned short* qbb   = (unsigned short*)alloc((size_t)TT * NHq * HDd * 2);
  unsigned short* kbb   = (unsigned short*)alloc((size_t)TT * NKVh * HDd * 2);
  unsigned short* vtb   = (unsigned short*)alloc((size_t)TT * NKVh * HDd * 2);
  unsigned short* attnb = (unsigned short*)alloc((size_t)TT * NHq * HDd * 2);
  float*          woout = (float*)alloc((size_t)TT * HH * 4);
  float*          x1    = (float*)alloc((size_t)TT * HH * 4);
  unsigned short* h2b   = (unsigned short*)alloc((size_t)TT * HH * 2);
  float*          lgt   = (float*)alloc((size_t)TT * EE * 4);
  float*          topw  = (float*)alloc((size_t)TT * TOPK * 4);
  int*            topi  = (int*)alloc((size_t)TT * TOPK * 4);
  int*            slot  = (int*)alloc((size_t)TT * TOPK * 4);
  int*            rowtk = (int*)alloc((size_t)TKR * 4);
  int*            cnts  = (int*)alloc(EE * 4);
  int*            offs  = (int*)alloc((EE + 1) * 4);
  int*            curs  = (int*)alloc(EE * 4);
  unsigned short* actb  = (unsigned short*)alloc((size_t)TKR * FF * 2);
  float*          ybuf  = (float*)alloc((size_t)TKR * HH * 4);
  // pre-transposed bf16 weights (N-major: [N][K])
  unsigned short* wqT = (unsigned short*)alloc((size_t)HH * NHq * HDd * 2);
  unsigned short* wkT = (unsigned short*)alloc((size_t)HH * NKVh * HDd * 2);
  unsigned short* wvT = (unsigned short*)alloc((size_t)HH * NKVh * HDd * 2);
  unsigned short* woT = (unsigned short*)alloc((size_t)NHq * HDd * HH * 2);
  unsigned short* gwT = (unsigned short*)alloc((size_t)HH * EE * 2);
  unsigned short* wiT = (unsigned short*)alloc((size_t)EE * HH * FF * 2);
  unsigned short* wuT = (unsigned short*)alloc((size_t)EE * HH * FF * 2);
  unsigned short* wdT = (unsigned short*)alloc((size_t)EE * FF * HH * 2);

  // ---- weight preprocessing: transpose + bf16 convert (one-time per launch) ----
  transpose_cvt_bf16<<<dim3((NHq * HDd) / 32, HH / 32, 1), 256, 0, stream>>>(wq, wqT, HH, NHq * HDd);
  transpose_cvt_bf16<<<dim3((NKVh * HDd) / 32, HH / 32, 1), 256, 0, stream>>>(wk, wkT, HH, NKVh * HDd);
  transpose_cvt_bf16<<<dim3((NKVh * HDd) / 32, HH / 32, 1), 256, 0, stream>>>(wv, wvT, HH, NKVh * HDd);
  transpose_cvt_bf16<<<dim3(HH / 32, (NHq * HDd) / 32, 1), 256, 0, stream>>>(wo, woT, NHq * HDd, HH);
  transpose_cvt_bf16<<<dim3(1, HH / 32, 1), 256, 0, stream>>>(gw, gwT, HH, EE);
  transpose_cvt_bf16<<<dim3(FF / 32, HH / 32, EE), 256, 0, stream>>>(wi, wiT, HH, FF);
  transpose_cvt_bf16<<<dim3(FF / 32, HH / 32, EE), 256, 0, stream>>>(wu, wuT, HH, FF);
  transpose_cvt_bf16<<<dim3(HH / 32, FF / 32, EE), 256, 0, stream>>>(wd, wdT, FF, HH);

  // ---- attention block ----
  rmsnorm_bf16<<<TT, 256, 0, stream>>>(x, ln1, h1b, HH);
  gemm_bf16<<<dim3((NHq * HDd) / 64, TT / 64), 32, 0, stream>>>(h1b, wqT, qf, TT, NHq * HDd, HH);
  gemm_bf16<<<dim3((NKVh * HDd) / 64, TT / 64), 32, 0, stream>>>(h1b, wkT, kf, TT, NKVh * HDd, HH);
  gemm_bf16<<<dim3((NKVh * HDd) / 64, TT / 64), 32, 0, stream>>>(h1b, wvT, vf, TT, NKVh * HDd, HH);
  qk_rope<<<dim3(TT, NHq + NKVh), 128, 0, stream>>>(qf, kf, qnw, knw, qbb, kbb);
  v_transpose<<<dim3(SS / 32, HDd / 32, BB * NKVh), 256, 0, stream>>>(vf, vtb);
  attn_fwd<<<dim3(SS / 16, NHq, BB), 32, 0, stream>>>(qbb, kbb, vtb, attnb);
  gemm_bf16<<<dim3(HH / 64, TT / 64), 32, 0, stream>>>(attnb, woT, woout, TT, HH, NHq * HDd);
  addvec<<<(TT * HH) / 256, 256, 0, stream>>>(x, woout, x1, TT * HH);

  // ---- MoE block ----
  rmsnorm_bf16<<<TT, 256, 0, stream>>>(x1, ln2, h2b, HH);
  gemm_bf16<<<dim3((EE + 63) / 64, TT / 64), 32, 0, stream>>>(h2b, gwT, lgt, TT, EE, HH);
  zero_i32<<<1, 64, 0, stream>>>(cnts, EE);
  router_topk<<<TT, 32, 0, stream>>>(lgt, topw, topi, cnts);
  scan_offsets<<<1, 32, 0, stream>>>(cnts, offs, curs);
  assign_slots<<<(TKR + 255) / 256, 256, 0, stream>>>(topi, curs, slot, rowtk);
  // fused gate/up + SwiGLU (A gathered through rowtok), then down projection
  moe_up_swiglu<<<dim3(FF / 32, TT / 64, EE), 32, 0, stream>>>(h2b, wiT, wuT, actb, offs, rowtk);
  gemm_moe_down<<<dim3(HH / 64, TT / 64, EE), 32, 0, stream>>>(actb, wdT, ybuf, offs, FF, HH);
  combine_moe<<<TT, 256, 0, stream>>>(x1, ybuf, topw, slot, out);
}
